// MoELayer_65446711656755
// MI455X (gfx1250) — compile-verified, matrix operand order checked by reading, not run on hardware
//
#include <hip/hip_runtime.h>
#include <math.h>
#include <stdint.h>

// ---- problem dims (fixed by reference) ----
#define B_SZ  4
#define S_SZ  2048
#define N_TOK (B_SZ * S_SZ)   // 8192 tokens
#define D_DIM 1024
#define H_DIM 4096
#define N_EXP 8

// ---- tiling ----
#define TM 128
#define TN 128
#define TK 32
#define KPAD 8
#define KST (TK + KPAD)        // 40 elems = 80 B row stride (16B aligned)

typedef __attribute__((ext_vector_type(16))) __bf16 v16bf;
typedef __attribute__((ext_vector_type(8)))  __bf16 v8bf;
typedef __attribute__((ext_vector_type(8)))  float  v8f;
typedef __attribute__((ext_vector_type(4)))  unsigned int v4u;
typedef __attribute__((ext_vector_type(8)))  int v8i;
typedef __attribute__((ext_vector_type(4)))  int v4i;

__device__ __forceinline__ __bf16 f2bf(float f) {
  unsigned u = __builtin_bit_cast(unsigned, f);
  u += 0x7fffu + ((u >> 16) & 1u);          // round-to-nearest-even
  unsigned short s = (unsigned short)(u >> 16);
  return __builtin_bit_cast(__bf16, s);
}

// ---- Tensor Data Mover: 2D bf16 tile -> LDS, with HW row padding ----
// D# per CDNA5 ISA. data_size=1 (2 bytes).
// tile0 = contiguous elems per row, tile1 = rows, stride0 = row stride (elems).
// pad: after every 16 DWORDs (=32 bf16, one tile row) insert 4 DWORDs (=8 bf16)
// -> LDS row stride becomes KST=40 bf16, matching the fragment loader.
__device__ __forceinline__ void tdm_load_2d(unsigned lds_addr, const void* gsrc,
                                            unsigned tile0, unsigned tile1,
                                            unsigned long long stride0) {
  unsigned long long ga = (unsigned long long)(uintptr_t)gsrc;
  v4u g0;
  g0.x = 1u;                                          // count=1, user-mode D#
  g0.y = lds_addr;                                    // lds_addr [63:32]
  g0.z = (unsigned)(ga & 0xffffffffu);                // global_addr[31:0]
  g0.w = (unsigned)((ga >> 32) & 0x1ffffffu)          // global_addr[56:32]
       | (2u << 30);                                  // type=2 ("image")
  const unsigned tdim0 = 1u << 20, tdim1 = 1u << 20;  // tiles never OOB
  v8i g1;
  g1[0] = (int)((1u << 16)                            // data_size = 2 bytes
              | (1u << 20)                            // pad_enable
              | (3u << 22)                            // pad_interval: 16 DWORDs
              | (3u << 25));                          // pad_amount: 4 DWORDs
  g1[1] = (int)((tdim0 & 0xffffu) << 16);             // tensor_dim0[15:0]
  g1[2] = (int)(((tdim0 >> 16) & 0xffffu) | ((tdim1 & 0xffffu) << 16));
  g1[3] = (int)(((tdim1 >> 16) & 0xffffu) | ((tile0 & 0xffffu) << 16));
  g1[4] = (int)(tile1 & 0xffffu);                     // tile_dim1 (tile_dim2=0)
  g1[5] = (int)(unsigned)(stride0 & 0xffffffffu);     // tensor_dim0_stride lo
  g1[6] = (int)(unsigned)((stride0 >> 32) & 0xffffu); // stride hi; dim1_stride=0
  g1[7] = 0;
  v4i z4 = {0, 0, 0, 0};
  v8i z8 = {0, 0, 0, 0, 0, 0, 0, 0};
  __builtin_amdgcn_tensor_load_to_lds(g0, g1, z4, z4, z8, 0);
}

// Load one 16x32 bf16 WMMA operand fragment from an LDS tile stored
// row-major ([row/col][k]) with stride KST. ISA packing (CDNA5 ISA):
// lanes 0-15 -> K 0..7 & 16..23, lanes 16-31 -> K 8..15 & 24..31.
__device__ __forceinline__ v16bf load_frag16x32(const __bf16* base) {
  int lane = threadIdx.x & 31;
  int r  = lane & 15;
  int kb = (lane >> 4) << 3;
  const __bf16* p = base + r * KST + kb;
  v8bf lo = *(const v8bf*)(p);
  v8bf hi = *(const v8bf*)(p + 16);
  return __builtin_shufflevector(lo, hi, 0,1,2,3,4,5,6,7,8,9,10,11,12,13,14,15);
}

// ---------------- zero output + counters ----------------
__global__ void moe_zero(float* __restrict__ out, int n, int* __restrict__ cnt) {
  int i = blockIdx.x * blockDim.x + threadIdx.x;
  if (i < n) out[i] = 0.0f;
  if (i < N_EXP) cnt[i] = 0;
}

// ---------------- x: fp32 -> bf16 (8 elems / thread) ----------------
__global__ __launch_bounds__(256)
void moe_cvt_x(const float* __restrict__ x, __bf16* __restrict__ xb, int n8) {
  int i = blockIdx.x * blockDim.x + threadIdx.x;
  if (i >= n8) return;
  const float* s = x + (size_t)i * 8;
  v8bf o;
#pragma unroll
  for (int j = 0; j < 8; ++j) o[j] = f2bf(s[j]);
  *(v8bf*)(xb + (size_t)i * 8) = o;
}

// ---- W: fp32 [R][C] -> bf16 [C][R] (transpose+convert, per expert z) ----
__global__ __launch_bounds__(256)
void moe_transcvt(const float* __restrict__ src, __bf16* __restrict__ dst,
                  int R, int C) {
  __shared__ __bf16 tile[32][33];
  const size_t eoff = (size_t)blockIdx.z * R * C;
  int c0 = blockIdx.x * 32, r0 = blockIdx.y * 32;
  int tx = threadIdx.x & 31, ty = threadIdx.x >> 5;   // ty: 0..7
#pragma unroll
  for (int i = 0; i < 32; i += 8)
    tile[ty + i][tx] = f2bf(src[eoff + (size_t)(r0 + ty + i) * C + c0 + tx]);
  __syncthreads();
#pragma unroll
  for (int i = 0; i < 32; i += 8)
    dst[eoff + (size_t)(c0 + ty + i) * R + r0 + tx] = tile[tx][ty + i];
}

// ---------------- gating: logits -> top2 -> softmax -> route ----------------
__global__ __launch_bounds__(256)
void moe_gate(const float* __restrict__ x, const float* __restrict__ Wg,
              const float* __restrict__ bg,
              int* __restrict__ cnt, int* __restrict__ tokIds,
              float* __restrict__ wts) {
  int n = blockIdx.x * blockDim.x + threadIdx.x;
  if (n >= N_TOK) return;

  float logit[N_EXP];
#pragma unroll
  for (int e = 0; e < N_EXP; ++e) logit[e] = bg[e];

  const float* xr = x + (size_t)n * D_DIM;
  for (int d = 0; d < D_DIM; ++d) {
    float xv = xr[d];
    const float* wr = Wg + (size_t)d * N_EXP;
#pragma unroll
    for (int e = 0; e < N_EXP; ++e) logit[e] += xv * wr[e];
  }

  int e0 = 0; float v0 = logit[0];
#pragma unroll
  for (int e = 1; e < N_EXP; ++e) if (logit[e] > v0) { v0 = logit[e]; e0 = e; }
  int e1 = (e0 == 0) ? 1 : 0; float v1 = logit[e1];
#pragma unroll
  for (int e = 0; e < N_EXP; ++e)
    if (e != e0 && e != e1 && logit[e] > v1) { v1 = logit[e]; e1 = e; }

  float ex = expf(v1 - v0);
  float inv = 1.0f / (1.0f + ex);
  float w0 = inv, w1 = ex * inv;

  int p0 = atomicAdd(&cnt[e0], 1);
  tokIds[e0 * N_TOK + p0] = n; wts[e0 * N_TOK + p0] = w0;
  int p1 = atomicAdd(&cnt[e1], 1);
  tokIds[e1 * N_TOK + p1] = n; wts[e1 * N_TOK + p1] = w1;
}

// ---------------- tiny exclusive scan for hbuf row offsets ----------------
__global__ void moe_scan(const int* __restrict__ cnt, int* __restrict__ offs) {
  if (blockIdx.x == 0 && threadIdx.x == 0) {
    int s = 0;
    for (int e = 0; e < N_EXP; ++e) { offs[e] = s; s += cnt[e]; }
  }
}

// ---------------- GEMM1: h = gelu(gather(xb) @ W1[e] + b1[e]) -> bf16 ------
// A: gathered token rows (vector LDS copies). B: TDM double-buffered, from
// pre-transposed w1t[e] = [H][D] so K is contiguous.
__global__ __launch_bounds__(256)
void moe_gemm1(const __bf16* __restrict__ xb, const __bf16* __restrict__ w1t,
               const float* __restrict__ b1,
               const int* __restrict__ cnt, const int* __restrict__ offs,
               const int* __restrict__ tokIds, __bf16* __restrict__ hbuf) {
  const int e  = blockIdx.z;
  const int M  = cnt[e];
  const int r0 = blockIdx.x * TM;
  if (r0 >= M) return;
  const int c0 = blockIdx.y * TN;

  __shared__ __align__(16) __bf16 As[TM][KST];
  __shared__ __align__(16) __bf16 Bs[2][TN][KST];
  __shared__ int toks[TM];

  const int t = threadIdx.x;
  if (t < TM) {
    int rr = r0 + t;
    toks[t] = tokIds[e * N_TOK + (rr < M ? rr : M - 1)];
  }
  __syncthreads();

  const __bf16* w1te = w1t + (size_t)e * H_DIM * D_DIM + (size_t)c0 * D_DIM;
  const unsigned bsA[2] = { (unsigned)(uintptr_t)&Bs[0][0][0],
                            (unsigned)(uintptr_t)&Bs[1][0][0] };
  const int wave = t >> 5, wm = wave >> 1, wn = wave & 1;
  const int NK = D_DIM / TK;

  if (t < 32) tdm_load_2d(bsA[0], w1te, TK, TN, D_DIM);   // prologue tile 0

  v8f acc[2][4] = {};

  for (int i = 0; i < NK; ++i) {
    const int kk = i * TK;
    __syncthreads();                 // all done reading As / Bs[(i-1)&1]
    if (t < 32) {
      if (i + 1 < NK) {
        tdm_load_2d(bsA[(i + 1) & 1], w1te + (size_t)(kk + TK), TK, TN, D_DIM);
        __builtin_amdgcn_s_wait_tensorcnt(1);   // tile i landed
      } else {
        __builtin_amdgcn_s_wait_tensorcnt(0);
      }
    }
    { // stage A: gathered bf16 rows, two 16B copies per thread
      int row = t >> 1, koff = (t & 1) * 16;
      const __bf16* src = xb + (size_t)toks[row] * D_DIM + kk + koff;
      *(v8bf*)&As[row][koff]     = *(const v8bf*)(src);
      *(v8bf*)&As[row][koff + 8] = *(const v8bf*)(src + 8);
    }
    __syncthreads();                 // publish As + Bs[i&1]

    v16bf af[2], bfr[4];
#pragma unroll
    for (int a = 0; a < 2; ++a) af[a] = load_frag16x32(&As[wm * 32 + a * 16][0]);
#pragma unroll
    for (int b = 0; b < 4; ++b) bfr[b] = load_frag16x32(&Bs[i & 1][wn * 64 + b * 16][0]);
#pragma unroll
    for (int a = 0; a < 2; ++a)
#pragma unroll
      for (int b = 0; b < 4; ++b)
        acc[a][b] = __builtin_amdgcn_wmma_f32_16x16x32_bf16(
            false, af[a], false, bfr[b], (short)0, acc[a][b], false, false);
  }

  // epilogue: +b1, exact gelu, bf16 h rows at offs[e]+row
  const int lane = t & 31;
  const int ln = lane & 15;
  const int lr = (lane >> 4) * 8;
  const int base = offs[e];
#pragma unroll
  for (int a = 0; a < 2; ++a)
#pragma unroll
    for (int b = 0; b < 4; ++b) {
      int col = c0 + wn * 64 + b * 16 + ln;
      float bias = b1[e * H_DIM + col];
#pragma unroll
      for (int v = 0; v < 8; ++v) {
        int row = r0 + wm * 32 + a * 16 + lr + v;
        if (row < M) {
          float val = acc[a][b][v] + bias;
          float g = 0.5f * val * (1.0f + erff(val * 0.70710678118654752f));
          hbuf[(size_t)(base + row) * H_DIM + col] = f2bf(g);
        }
      }
    }
}

// ---------------- GEMM2: out += w * (h @ W2[e] + b2[e]) --------------------
// A rows are CONTIGUOUS in hbuf -> both A and B staged by TDM, double-buffered.
__global__ __launch_bounds__(256)
void moe_gemm2(const __bf16* __restrict__ hbuf, const __bf16* __restrict__ w2t,
               const float* __restrict__ b2,
               const int* __restrict__ cnt, const int* __restrict__ offs,
               const int* __restrict__ tokIds, const float* __restrict__ wts,
               float* __restrict__ out) {
  const int e  = blockIdx.z;
  const int M  = cnt[e];
  const int r0 = blockIdx.x * TM;
  if (r0 >= M) return;
  const int c0 = blockIdx.y * TN;

  __shared__ __align__(16) __bf16 As[2][TM][KST];
  __shared__ __align__(16) __bf16 Bs[2][TN][KST];
  __shared__ int   toks[TM];
  __shared__ float wrow[TM];

  const int t = threadIdx.x;
  if (t < TM) {
    int rr = r0 + t;
    int idx = e * N_TOK + (rr < M ? rr : M - 1);
    toks[t] = tokIds[idx];
    wrow[t] = wts[idx];
  }
  __syncthreads();

  const int base = offs[e];
  const __bf16* hsrc = hbuf + (size_t)(base + r0) * H_DIM;
  const __bf16* w2te = w2t + (size_t)e * D_DIM * H_DIM + (size_t)c0 * H_DIM;
  const unsigned asA[2] = { (unsigned)(uintptr_t)&As[0][0][0],
                            (unsigned)(uintptr_t)&As[1][0][0] };
  const unsigned bsA[2] = { (unsigned)(uintptr_t)&Bs[0][0][0],
                            (unsigned)(uintptr_t)&Bs[1][0][0] };
  const int wave = t >> 5, wm = wave >> 1, wn = wave & 1;
  const int NK = H_DIM / TK;

  if (t < 32) {                       // prologue tile 0 (A + B)
    tdm_load_2d(asA[0], hsrc, TK, TM, H_DIM);
    tdm_load_2d(bsA[0], w2te, TK, TN, H_DIM);
  }

  v8f acc[2][4] = {};

  for (int i = 0; i < NK; ++i) {
    const int kk = i * TK;
    __syncthreads();
    if (t < 32) {
      if (i + 1 < NK) {
        tdm_load_2d(asA[(i + 1) & 1], hsrc + (size_t)(kk + TK), TK, TM, H_DIM);
        tdm_load_2d(bsA[(i + 1) & 1], w2te + (size_t)(kk + TK), TK, TN, H_DIM);
        __builtin_amdgcn_s_wait_tensorcnt(2);   // pair i landed
      } else {
        __builtin_amdgcn_s_wait_tensorcnt(0);
      }
    }
    __syncthreads();

    v16bf af[2], bfr[4];
#pragma unroll
    for (int a = 0; a < 2; ++a) af[a] = load_frag16x32(&As[i & 1][wm * 32 + a * 16][0]);
#pragma unroll
    for (int b = 0; b < 4; ++b) bfr[b] = load_frag16x32(&Bs[i & 1][wn * 64 + b * 16][0]);
#pragma unroll
    for (int a = 0; a < 2; ++a)
#pragma unroll
      for (int b = 0; b < 4; ++b)
        acc[a][b] = __builtin_amdgcn_wmma_f32_16x16x32_bf16(
            false, af[a], false, bfr[b], (short)0, acc[a][b], false, false);
  }

  // epilogue: scatter-add weighted (y + b2); exactly 2 contributions/element
  const int lane = t & 31;
  const int ln = lane & 15;
  const int lr = (lane >> 4) * 8;
#pragma unroll
  for (int a = 0; a < 2; ++a)
#pragma unroll
    for (int b = 0; b < 4; ++b) {
      int col = c0 + wn * 64 + b * 16 + ln;
      float bias = b2[e * D_DIM + col];
#pragma unroll
      for (int v = 0; v < 8; ++v) {
        int rloc = wm * 32 + a * 16 + lr + v;
        int row = r0 + rloc;
        if (row < M) {
          float contrib = wrow[rloc] * (acc[a][b][v] + bias);
          atomicAdd(&out[(size_t)toks[rloc] * D_DIM + col], contrib);
        }
      }
    }
}

extern "C" void kernel_launch(void* const* d_in, const int* in_sizes, int n_in,
                              void* d_out, int out_size, void* d_ws, size_t ws_size,
                              hipStream_t stream) {
  (void)in_sizes; (void)n_in; (void)out_size; (void)ws_size;
  const float* x  = (const float*)d_in[0];
  const float* Wg = (const float*)d_in[1];
  const float* bg = (const float*)d_in[2];
  const float* W1 = (const float*)d_in[3];
  const float* b1 = (const float*)d_in[4];
  const float* W2 = (const float*)d_in[5];
  const float* b2 = (const float*)d_in[6];
  float* out = (float*)d_out;

  // workspace layout
  char* p = (char*)d_ws;
  int*    cnt    = (int*)p;    p += 256;
  int*    offs   = (int*)p;    p += 256;
  int*    tokIds = (int*)p;    p += (size_t)N_EXP * N_TOK * sizeof(int);
  float*  wts    = (float*)p;  p += (size_t)N_EXP * N_TOK * sizeof(float);
  __bf16* xb     = (__bf16*)p; p += (size_t)N_TOK * D_DIM * 2;          // 16 MiB
  __bf16* w1t    = (__bf16*)p; p += (size_t)N_EXP * H_DIM * D_DIM * 2;  // 64 MiB
  __bf16* w2t    = (__bf16*)p; p += (size_t)N_EXP * D_DIM * H_DIM * 2;  // 64 MiB
  __bf16* hbuf   = (__bf16*)p; // (2*N_TOK + TM) x H_DIM bf16, ~129 MiB

  int total = N_TOK * D_DIM;
  moe_zero<<<(total + 255) / 256, 256, 0, stream>>>(out, total, cnt);
  moe_cvt_x<<<(total / 8 + 255) / 256, 256, 0, stream>>>(x, xb, total / 8);
  // W1 [D][H] -> w1t [H][D];  W2 [H][D] -> w2t [D][H]   (bf16)
  moe_transcvt<<<dim3(H_DIM / 32, D_DIM / 32, N_EXP), 256, 0, stream>>>(W1, w1t, D_DIM, H_DIM);
  moe_transcvt<<<dim3(D_DIM / 32, H_DIM / 32, N_EXP), 256, 0, stream>>>(W2, w2t, H_DIM, D_DIM);
  moe_gate<<<N_TOK / 256, 256, 0, stream>>>(x, Wg, bg, cnt, tokIds, wts);
  moe_scan<<<1, 32, 0, stream>>>(cnt, offs);
  moe_gemm1<<<dim3(N_TOK / TM, H_DIM / TN, N_EXP), 256, 0, stream>>>(
      xb, w1t, b1, cnt, offs, tokIds, hbuf);
  moe_gemm2<<<dim3(N_TOK / TM, D_DIM / TN, N_EXP), 256, 0, stream>>>(
      hbuf, w2t, b2, cnt, offs, tokIds, wts, out);
}
